// _SetAbstraction_86165633892448
// MI455X (gfx1250) — matrix operand order repaired and checked by reading, not compiled, not run on hardware
//
#include <hip/hip_runtime.h>
#include <math.h>
#include <float.h>

#define BATCH   16
#define NPTS    4096
#define DFEAT   64
#define NPOINT  1024
#define NSAMPLE 32
#define CK0     96     // layer-0 in-channels padded: 3 (rel xyz) + 64 (feat) -> 96
#define CK1     64

typedef __attribute__((ext_vector_type(16))) __bf16 v16bf;
typedef __attribute__((ext_vector_type(8)))  __bf16 v8bf;
typedef __attribute__((ext_vector_type(8)))  float  v8f;

static __device__ __forceinline__ v8f v8f_zero() {
  v8f z = {0.f,0.f,0.f,0.f,0.f,0.f,0.f,0.f};
  return z;
}

static __device__ __forceinline__ v8f wmma_bf16(v16bf a, v16bf b, v8f c) {
  // D = A(16x32 bf16) x B(32x16 bf16) + C(16x16 f32)
  return __builtin_amdgcn_wmma_f32_16x16x32_bf16(false, a, false, b, (short)0, c,
                                                 false, false);
}

// A fragment (16x32 bf16, rows row0..row0+15, channels k0..k0+31) from LDS.
// ISA layout: lanes 0-15 row M=lane hold K {0..7, 16..23}; lanes 16-31 hold
// K {8..15, 24..31} -> two 16B ds_load_b128 per lane.
static __device__ __forceinline__ v16bf load_afrag(const __bf16* As, int ck,
                                                   int row0, int k0, int lane) {
  const int half = lane >> 4;
  const int r    = lane & 15;
  const __bf16* p = As + (row0 + r) * ck + k0 + half * 8;
  v8bf lo = *(const v8bf*)p;
  v8bf hi = *(const v8bf*)(p + 16);
  return __builtin_shufflevector(lo, hi, 0,1,2,3,4,5,6,7,8,9,10,11,12,13,14,15);
}

// B fragment (32x16 KxN) from packed bf16 weights Wp[outC][ck].
// ISA layout: lanes 0-15 col N=lane hold K {0..7, 16..23}; lanes 16-31 hold
// K {8..15, 24..31} -> two 16B loads per lane.
static __device__ __forceinline__ v16bf load_bfrag(const __bf16* Wp, int ck,
                                                   int n0, int k0, int lane) {
  const int half = lane >> 4;
  const int n    = n0 + (lane & 15);
  const __bf16* p = Wp + n * ck + k0 + half * 8;
  v8bf lo = *(const v8bf*)p;
  v8bf hi = *(const v8bf*)(p + 16);
  return __builtin_shufflevector(lo, hi, 0,1,2,3,4,5,6,7,8,9,10,11,12,13,14,15);
}

template<int NJ, int NK>
static __device__ __forceinline__ void conv_layer(const __bf16* As, int ck,
                                                  const __bf16* Wp,
                                                  v8f (&acc)[2][NJ], int lane) {
#pragma unroll
  for (int t = 0; t < 2; ++t)
#pragma unroll
    for (int j = 0; j < NJ; ++j) acc[t][j] = v8f_zero();
#pragma unroll
  for (int kc = 0; kc < NK; ++kc) {
    v16bf a0 = load_afrag(As, ck, 0,  kc * 32, lane);
    v16bf a1 = load_afrag(As, ck, 16, kc * 32, lane);
#pragma unroll
    for (int j = 0; j < NJ; ++j) {
      v16bf bf = load_bfrag(Wp, ck, j * 16, kc * 32, lane);
      acc[0][j] = wmma_bf16(a0, bf, acc[0][j]);
      acc[1][j] = wmma_bf16(a1, bf, acc[1][j]);
    }
  }
}

// C/D layout: channel = n0 + lane%16; rows t*16 + (lane>=16?8:0) + vgpr-index.
template<int NJ>
static __device__ __forceinline__ void do_stats(v8f (&acc)[2][NJ],
                                                const float* bias,
                                                float* sumP, float* sqP, int lane) {
  const int cl = lane & 15;
#pragma unroll
  for (int j = 0; j < NJ; ++j) {
    const int c = j * 16 + cl;
    const float bi = bias[c];
    float s = 0.f, q = 0.f;
#pragma unroll
    for (int t = 0; t < 2; ++t)
#pragma unroll
      for (int i = 0; i < 8; ++i) {
        const float v = acc[t][j][i] + bi;
        s += v;
        q = fmaf(v, v, q);
      }
    // lanes l and l+16 see the same channel: merge, then one atomic per channel
    s += __shfl_xor(s, 16);
    q += __shfl_xor(q, 16);
    if (lane < 16) {
      atomicAdd(&sumP[c], s);
      atomicAdd(&sqP[c], q);
    }
  }
}

template<int NJ>
static __device__ __forceinline__ void bn_relu_store(v8f (&acc)[2][NJ],
                                                     const float* bias,
                                                     const float* aP, const float* bP,
                                                     __bf16* Anext, int ckN, int lane) {
  const int cl   = lane & 15;
  const int half = lane >> 4;
#pragma unroll
  for (int j = 0; j < NJ; ++j) {
    const int c    = j * 16 + cl;
    const float sc = aP[c];
    const float sh = fmaf(sc, bias[c], bP[c]);   // fold conv bias into shift
#pragma unroll
    for (int t = 0; t < 2; ++t)
#pragma unroll
      for (int i = 0; i < 8; ++i) {
        float h = fmaf(acc[t][j][i], sc, sh);
        h = fmaxf(h, 0.f);
        Anext[(t * 16 + half * 8 + i) * ckN + c] = (__bf16)h;
      }
  }
}

// One wave per (b, m) group: gather 32x96 tile, run the 3-layer MLP with WMMA.
// STAGE 0/1/2: accumulate global BN stats for that layer and stop.
// STAGE 3: full pipeline + max-pool over the 32 samples -> new_points.
template<int STAGE>
__global__ __launch_bounds__(32) void mlp_kernel(
    const float* __restrict__ points, const float* __restrict__ xyz,
    const float* __restrict__ nxyz, const int* __restrict__ knn,
    const __bf16* __restrict__ W0p, const __bf16* __restrict__ W1p,
    const __bf16* __restrict__ W2p,
    const float* __restrict__ cb0, const float* __restrict__ cb1,
    const float* __restrict__ cb2,
    const float* __restrict__ ab, float* __restrict__ stats,
    float* __restrict__ out_pts) {
  __shared__ __attribute__((aligned(32))) __bf16 As[32 * CK0];

  const int gid  = blockIdx.x;          // b*1024 + m
  const int b    = gid >> 10;
  const int lane = threadIdx.x;

  __builtin_prefetch(W0p, 0, 1);        // global_prefetch_b8: warm weights

  // ---- gather: lane s builds row s = [rel_xyz(3) | feat(64) | pad(29)] ----
  {
    const int s    = lane;
    const int pidx = knn[(long)gid * NSAMPLE + s];
    const float cx = nxyz[(long)gid * 3 + 0];
    const float cy = nxyz[(long)gid * 3 + 1];
    const float cz = nxyz[(long)gid * 3 + 2];
    const float* X = xyz    + ((long)b * NPTS + pidx) * 3;
    const float* P = points + ((long)b * NPTS + pidx) * DFEAT;
    __bf16* row = As + s * CK0;
    row[0] = (__bf16)(X[0] - cx);
    row[1] = (__bf16)(X[1] - cy);
    row[2] = (__bf16)(X[2] - cz);
#pragma unroll
    for (int c2 = 0; c2 < DFEAT; c2 += 4) {
      const float4 v = *(const float4*)(P + c2);
      row[3 + c2 + 0] = (__bf16)v.x;
      row[3 + c2 + 1] = (__bf16)v.y;
      row[3 + c2 + 2] = (__bf16)v.z;
      row[3 + c2 + 3] = (__bf16)v.w;
    }
#pragma unroll
    for (int c2 = 3 + DFEAT; c2 < CK0; ++c2) row[c2] = (__bf16)0.f;
  }
  __syncthreads();

  // ---- layer 0: 96 -> 64 ----
  v8f acc[2][4];
  conv_layer<4, 3>(As, CK0, W0p, acc, lane);
  if (STAGE == 0) { do_stats<4>(acc, cb0, stats + 0, stats + 64, lane); return; }
  __syncthreads();
  bn_relu_store<4>(acc, cb0, ab + 0, ab + 64, As, CK1, lane);
  __syncthreads();

  // ---- layer 1: 64 -> 64 ----
  conv_layer<4, 2>(As, CK1, W1p, acc, lane);
  if (STAGE == 1) { do_stats<4>(acc, cb1, stats + 128, stats + 192, lane); return; }
  __syncthreads();
  bn_relu_store<4>(acc, cb1, ab + 128, ab + 192, As, CK1, lane);
  __syncthreads();

  // ---- layer 2: 64 -> 128 ----
  v8f acc2[2][8];
  conv_layer<8, 2>(As, CK1, W2p, acc2, lane);
  if (STAGE == 2) { do_stats<8>(acc2, cb2, stats + 256, stats + 384, lane); return; }

  // ---- final: BN2 + ReLU + max over 32 samples ----
  const int cl = lane & 15;
#pragma unroll
  for (int j = 0; j < 8; ++j) {
    const int c    = j * 16 + cl;
    const float sc = ab[256 + c];
    const float sh = fmaf(sc, cb2[c], ab[384 + c]);
    float m = -FLT_MAX;
#pragma unroll
    for (int t = 0; t < 2; ++t)
#pragma unroll
      for (int i = 0; i < 8; ++i) {
        float h = fmaf(acc2[t][j][i], sc, sh);
        h = fmaxf(h, 0.f);
        m = fmaxf(m, h);
      }
    m = fmaxf(m, __shfl_xor(m, 16));   // merge the two 8-row lane halves
    if (lane < 16) out_pts[(long)gid * 128 + c] = m;
  }
}

// ---- sequential farthest point sampling: one block per batch ----
__global__ __launch_bounds__(1024) void fps_kernel(
    const float* __restrict__ xyz, int* __restrict__ fpsIdx,
    float* __restrict__ nxyz, float* __restrict__ out_xyz,
    int* __restrict__ out_idx) {
  const int b   = blockIdx.x;
  const int tid = threadIdx.x;
  __shared__ float s_val[32];
  __shared__ int   s_idx[32];
  __shared__ int   s_far;
  __shared__ int   seq[NPOINT];

  const float* X = xyz + (long)b * NPTS * 3;
  float px[4], py[4], pz[4], dst[4];
#pragma unroll
  for (int j = 0; j < 4; ++j) {
    const int p = tid + j * 1024;
    px[j] = X[p * 3 + 0];
    py[j] = X[p * 3 + 1];
    pz[j] = X[p * 3 + 2];
    dst[j] = FLT_MAX;
  }

  int far = NPTS / 2;
  for (int it = 0; it < NPOINT; ++it) {
    if (tid == 0) seq[it] = far;
    const float cx = X[far * 3 + 0];
    const float cy = X[far * 3 + 1];
    const float cz = X[far * 3 + 2];
    float bv = -1.f;
    int   bi = 0;
#pragma unroll
    for (int j = 0; j < 4; ++j) {
      const float dx = px[j] - cx, dy = py[j] - cy, dz = pz[j] - cz;
      const float dd = fmaf(dx, dx, fmaf(dy, dy, dz * dz));
      dst[j] = fminf(dst[j], dd);
      const int p = tid + j * 1024;
      if (dst[j] > bv || (dst[j] == bv && p < bi)) { bv = dst[j]; bi = p; }
    }
    // wave32 argmax (first-occurrence tie-break)
#pragma unroll
    for (int off = 16; off > 0; off >>= 1) {
      const float ov = __shfl_xor(bv, off);
      const int   oi = __shfl_xor(bi, off);
      if (ov > bv || (ov == bv && oi < bi)) { bv = ov; bi = oi; }
    }
    const int w = tid >> 5;
    if ((tid & 31) == 0) { s_val[w] = bv; s_idx[w] = bi; }
    __syncthreads();
    if (tid < 32) {
      float v = s_val[tid];
      int   i2 = s_idx[tid];
#pragma unroll
      for (int off = 16; off > 0; off >>= 1) {
        const float ov = __shfl_xor(v, off);
        const int   oi = __shfl_xor(i2, off);
        if (ov > v || (ov == v && oi < i2)) { v = ov; i2 = oi; }
      }
      if (tid == 0) s_far = i2;
    }
    __syncthreads();
    far = s_far;
  }

  // emit fps_idx and new_xyz (tid == m)
  if (tid < NPOINT) {
    const int id = seq[tid];
    fpsIdx[b * NPOINT + tid]  = id;
    out_idx[b * NPOINT + tid] = id;
    const float x0 = X[id * 3 + 0], y0 = X[id * 3 + 1], z0 = X[id * 3 + 2];
    const long o = ((long)b * NPOINT + tid) * 3;
    nxyz[o + 0] = x0; nxyz[o + 1] = y0; nxyz[o + 2] = z0;
    out_xyz[o + 0] = x0; out_xyz[o + 1] = y0; out_xyz[o + 2] = z0;
  }
}

// ---- kNN: one workgroup per query; LDS distance array + 32 argmin rounds ----
__global__ __launch_bounds__(256) void topk_kernel(
    const float* __restrict__ xyz, const float* __restrict__ nxyz,
    int* __restrict__ knn) {
  const int gid = blockIdx.x;       // b*1024 + m
  const int b   = gid >> 10;
  const int tid = threadIdx.x;
  __shared__ float dist[NPTS];
  __shared__ float rv[256];
  __shared__ int   ri[256];

  const float* X = xyz + (long)b * NPTS * 3;
  const float cx = nxyz[(long)gid * 3 + 0];
  const float cy = nxyz[(long)gid * 3 + 1];
  const float cz = nxyz[(long)gid * 3 + 2];
  for (int p = tid; p < NPTS; p += 256) {
    const float dx = X[p * 3 + 0] - cx;
    const float dy = X[p * 3 + 1] - cy;
    const float dz = X[p * 3 + 2] - cz;
    dist[p] = fmaf(dx, dx, fmaf(dy, dy, dz * dz));
  }
  __syncthreads();

  for (int k = 0; k < NSAMPLE; ++k) {
    float bv = FLT_MAX;
    int   bi = NPTS;
    for (int p = tid; p < NPTS; p += 256) {
      const float v = dist[p];
      if (v < bv || (v == bv && p < bi)) { bv = v; bi = p; }
    }
    rv[tid] = bv; ri[tid] = bi;
    __syncthreads();
    for (int off = 128; off > 0; off >>= 1) {
      if (tid < off) {
        const float ov = rv[tid + off];
        const int   oi = ri[tid + off];
        if (ov < rv[tid] || (ov == rv[tid] && oi < ri[tid])) {
          rv[tid] = ov; ri[tid] = oi;
        }
      }
      __syncthreads();
    }
    if (tid == 0) {
      knn[(long)gid * NSAMPLE + k] = ri[0];
      dist[ri[0]] = FLT_MAX;        // remove selected
    }
    __syncthreads();
  }
}

__global__ void zero_stats_kernel(float* __restrict__ s) {
  s[threadIdx.x] = 0.f;             // 512 floats: sum/sq for 3 layers
}

__global__ void prep_weights_kernel(const float* __restrict__ W0,
                                    const float* __restrict__ W1,
                                    const float* __restrict__ W2,
                                    __bf16* __restrict__ W0p,
                                    __bf16* __restrict__ W1p,
                                    __bf16* __restrict__ W2p) {
  const int t = blockIdx.x * blockDim.x + threadIdx.x;
  if (t < 64 * CK0) {
    const int r = t / CK0, c = t % CK0;
    W0p[t] = (__bf16)(c < 67 ? W0[r * 67 + c] : 0.f);
  } else if (t < 64 * CK0 + 64 * 64) {
    const int u = t - 64 * CK0;
    W1p[u] = (__bf16)W1[u];
  } else if (t < 64 * CK0 + 64 * 64 + 128 * 64) {
    const int u = t - 64 * CK0 - 64 * 64;
    W2p[u] = (__bf16)W2[u];
  }
}

__global__ void finalize_bn_kernel(const float* __restrict__ sum,
                                   const float* __restrict__ sq,
                                   const float* __restrict__ gamma,
                                   const float* __restrict__ beta,
                                   float* __restrict__ a,
                                   float* __restrict__ bshift, int C) {
  const int c = threadIdx.x;
  if (c >= C) return;
  const float cnt = (float)(BATCH * NPOINT * NSAMPLE);
  const float mu  = sum[c] / cnt;
  const float var = sq[c] / cnt - mu * mu;
  const float s   = gamma[c] * rsqrtf(var + 1e-5f);
  a[c]      = s;
  bshift[c] = beta[c] - mu * s;
}

extern "C" void kernel_launch(void* const* d_in, const int* in_sizes, int n_in,
                              void* d_out, int out_size, void* d_ws, size_t ws_size,
                              hipStream_t stream) {
  const float* xyz    = (const float*)d_in[0];
  const float* points = (const float*)d_in[1];
  const float* W0  = (const float*)d_in[2];
  const float* b0  = (const float*)d_in[3];
  const float* g0  = (const float*)d_in[4];
  const float* be0 = (const float*)d_in[5];
  const float* W1  = (const float*)d_in[6];
  const float* b1  = (const float*)d_in[7];
  const float* g1  = (const float*)d_in[8];
  const float* be1 = (const float*)d_in[9];
  const float* W2  = (const float*)d_in[10];
  const float* b2  = (const float*)d_in[11];
  const float* g2  = (const float*)d_in[12];
  const float* be2 = (const float*)d_in[13];

  // outputs: new_xyz [16,1024,3] | new_points [16,1024,128] | fps_idx [16,1024]
  float* out_f   = (float*)d_out;
  float* out_xyz = out_f;
  float* out_pts = out_f + BATCH * NPOINT * 3;
  int*   out_idx = (int*)(out_f + BATCH * NPOINT * 3 + BATCH * NPOINT * 128);

  // workspace layout (bytes)
  char* ws = (char*)d_ws;
  float*  stats = (float*)(ws + 0);        // 512 f: sum0,sq0,sum1,sq1,sum2,sq2
  float*  ab    = (float*)(ws + 2048);     // 512 f: a0,b0,a1,b1,a2,b2
  float*  nxyz  = (float*)(ws + 4096);     // 49152 f
  int*    fpsI  = (int*)(ws + 200704);     // 16384 i
  int*    knn   = (int*)(ws + 266240);     // 524288 i
  __bf16* W0p   = (__bf16*)(ws + 2363392); // 64x96 bf16 (32B aligned)
  __bf16* W1p   = W0p + 64 * CK0;          // 64x64
  __bf16* W2p   = W1p + 64 * 64;           // 128x64

  const int ngroups = BATCH * NPOINT;

  zero_stats_kernel<<<1, 512, 0, stream>>>(stats);
  prep_weights_kernel<<<72, 256, 0, stream>>>(W0, W1, W2, W0p, W1p, W2p);
  fps_kernel<<<BATCH, 1024, 0, stream>>>(xyz, fpsI, nxyz, out_xyz, out_idx);
  topk_kernel<<<ngroups, 256, 0, stream>>>(xyz, nxyz, knn);

  mlp_kernel<0><<<ngroups, 32, 0, stream>>>(points, xyz, nxyz, knn, W0p, W1p, W2p,
                                            b0, b1, b2, ab, stats, out_pts);
  finalize_bn_kernel<<<1, 64, 0, stream>>>(stats + 0, stats + 64, g0, be0,
                                           ab + 0, ab + 64, 64);
  mlp_kernel<1><<<ngroups, 32, 0, stream>>>(points, xyz, nxyz, knn, W0p, W1p, W2p,
                                            b0, b1, b2, ab, stats, out_pts);
  finalize_bn_kernel<<<1, 64, 0, stream>>>(stats + 128, stats + 192, g1, be1,
                                           ab + 128, ab + 192, 64);
  mlp_kernel<2><<<ngroups, 32, 0, stream>>>(points, xyz, nxyz, knn, W0p, W1p, W2p,
                                            b0, b1, b2, ab, stats, out_pts);
  finalize_bn_kernel<<<1, 128, 0, stream>>>(stats + 256, stats + 384, g2, be2,
                                            ab + 256, ab + 384, 128);
  mlp_kernel<3><<<ngroups, 32, 0, stream>>>(points, xyz, nxyz, knn, W0p, W1p, W2p,
                                            b0, b1, b2, ab, stats, out_pts);
}
